// JIIF_27736898797755
// MI455X (gfx1250) — compile-verified, matrix operand order checked by reading, not run on hardware
//
#include <hip/hip_runtime.h>
#include <hip/hip_bf16.h>

// ---------------------------------------------------------------------------
// JIIF fused gather + MLP for MI455X (gfx1250), bf16 WMMA path.
// Rows = B*N*4 = 1,048,576 ; MLP 386->1024->512->256->128->2 fused in LDS.
// Round 2: each wave computes BOTH 16-row tiles against one streamed
// B-fragment (2 accumulators) -> every weight byte read once per WG,
// halving L2 weight traffic per WMMA.
// ---------------------------------------------------------------------------

typedef __attribute__((ext_vector_type(8)))  float          f32x8;
typedef __attribute__((ext_vector_type(16))) __bf16         bf16x16;
typedef __attribute__((ext_vector_type(8)))  unsigned short u16x8;

union FragU { u16x8 h[2]; bf16x16 v; };

#define CH      128      // feat channels
#define GD      128      // guide channels
#define IN_DIM  386
#define K0PAD   416      // 13 * 32
#define TILE_M  32
#define NQ      65536
#define LDS_BYTES 98304  // bufA 32KB + bufB 64KB

// half-element offsets of packed bf16 weight blobs in workspace
#define OFF_W0  0
#define OFF_W1  425984   // 416*1024
#define OFF_W2  950272   // + 1024*512
#define OFF_W3  1081344  // + 512*256
#define OFF_PRED_BYTES 2228224  // (OFF_W3 + 256*128) * 2

__device__ __forceinline__ unsigned short f2bf(float x) {
  unsigned u = __builtin_bit_cast(unsigned, x);
  u += 0x7FFFu + ((u >> 16) & 1u);               // round-to-nearest-even
  return (unsigned short)(u >> 16);
}
__device__ __forceinline__ float bf2f(unsigned short h) {
  unsigned u = ((unsigned)h) << 16;
  return __builtin_bit_cast(float, u);
}

// ---------------------------------------------------------------------------
// Pack one f32 weight matrix [Kreal x N] into fragment-major bf16 layout.
// Fragment (ct,kt): 32 lanes x 16 halves; lane half j holds
//   k = kt*32 + (j&7) + ((j>>3)<<4) + ((lane>>4)<<3),  n = ct*16 + (lane&15)
// matching the ISA 16-bit 32x16 fragment striping.
// ---------------------------------------------------------------------------
__global__ __launch_bounds__(256) void pack_w_kernel(
    const float* __restrict__ W, int Kreal, int nkt, int N,
    unsigned short* __restrict__ out)
{
  const int tid  = blockIdx.x * 256 + threadIdx.x;
  const int j    = tid & 15;
  const int lane = (tid >> 4) & 31;
  const int f    = tid >> 9;
  const int kt   = f % nkt;
  const int ct   = f / nkt;
  const int klocal = (j & 7) + ((j >> 3) << 4) + ((lane >> 4) << 3);
  const int k = kt * 32 + klocal;
  const int n = ct * 16 + (lane & 15);
  const float v = (k < Kreal) ? W[(size_t)k * N + n] : 0.0f;
  out[tid] = f2bf(v);
}

// ---------------------------------------------------------------------------
// One dense layer: LDS bf16 [32 x K] @ packed-bf16 [K x Nout] -> LDS bf16
// [32 x Nout], bias + optional ReLU.
// 8 waves each own column tiles ct = wave, wave+8, ... ; per B-fragment
// (2x global_load_b128, L2-resident) a wave issues TWO WMMAs — one per
// 16-row tile — so each weight byte is fetched exactly once per workgroup.
// A-fragments come from LDS (2x ds_load_b128 per tile) per the ISA 16-bit
// A striping: lanes 0-15 hold K 0-7/16-23, lanes 16-31 hold K 8-15/24-31.
// ---------------------------------------------------------------------------
__device__ __forceinline__ void layer_gemm(
    const unsigned short* __restrict__ src, int K,
    unsigned short* __restrict__ dst, int Nout,
    const unsigned short* __restrict__ wp,
    const float* __restrict__ bias, bool relu,
    int lane, int wave)
{
  const int nkt  = K >> 5;
  const int nnt  = Nout >> 4;
  const int nlo  = lane & 15;
  const int msub = (lane >> 4) << 3;               // 0 or 8: lane's row sub-base
  const unsigned short* arow0 = src + nlo * K + msub;        // row tile 0
  const unsigned short* arow1 = src + (16 + nlo) * K + msub; // row tile 1

  for (int ct = wave; ct < nnt; ct += 8) {
    f32x8 acc0 = {};
    f32x8 acc1 = {};
    const unsigned short* ap0 = arow0;
    const unsigned short* ap1 = arow1;
    const unsigned short* bp = wp + ((size_t)(ct * nkt) * 32 + lane) * 16;
    for (int kt = 0; kt < nkt; ++kt) {
      FragU a0, a1, b;
      b.h[0]  = *(const u16x8*)(bp);
      b.h[1]  = *(const u16x8*)(bp + 8);
      a0.h[0] = *(const u16x8*)(ap0);
      a0.h[1] = *(const u16x8*)(ap0 + 16);
      a1.h[0] = *(const u16x8*)(ap1);
      a1.h[1] = *(const u16x8*)(ap1 + 16);
      acc0 = __builtin_amdgcn_wmma_f32_16x16x32_bf16(
          false, a0.v, false, b.v, (short)0, acc0, false, false);
      acc1 = __builtin_amdgcn_wmma_f32_16x16x32_bf16(
          false, a1.v, false, b.v, (short)0, acc1, false, false);
      ap0 += 32;
      ap1 += 32;
      bp  += 512;
    }
    const float bv = bias[ct * 16 + nlo];
    unsigned short* d0 = dst + msub * Nout + ct * 16 + nlo;          // rows msub..+7
    unsigned short* d1 = dst + (16 + msub) * Nout + ct * 16 + nlo;   // rows 16+msub..
#pragma unroll
    for (int i = 0; i < 8; ++i) {
      float x0 = acc0[i] + bv;
      float x1 = acc1[i] + bv;
      if (relu) { x0 = fmaxf(x0, 0.0f); x1 = fmaxf(x1, 0.0f); }
      d0[i * Nout] = f2bf(x0);
      d1[i * Nout] = f2bf(x1);
    }
  }
}

// ---------------------------------------------------------------------------
// Fused gather + MLP.  One workgroup = 32 rows of the (B*N*4)-row GEMM.
// Row g: shift s = g&3, query bn = g>>2, batch b = bn>>16.
// ---------------------------------------------------------------------------
__global__ __launch_bounds__(256) void jiif_mlp_kernel(
    const float* __restrict__ feat, const float* __restrict__ coord,
    const float* __restrict__ hr_guide, const float* __restrict__ lr_guide,
    const unsigned short* __restrict__ wpack,
    const float* __restrict__ b0, const float* __restrict__ b1,
    const float* __restrict__ b2, const float* __restrict__ b3,
    const float* __restrict__ W4, const float* __restrict__ b4,
    float* __restrict__ preds)
{
  extern __shared__ __align__(16) unsigned short smem[];
  unsigned short* bufA = smem;           // 16384 halves (input/act2/act4)
  unsigned short* bufB = smem + 16384;   // 32768 halves (act1/act3)

  const int t = threadIdx.x;
  const int lane = t & 31, wave = t >> 5;

  // ---- gather: build 32 x 416 bf16 input tile (8 threads per row) ----
  {
    const int r = t >> 3;
    const int q = t & 7;
    const int g  = blockIdx.x * TILE_M + r;
    const int s  = g & 3;
    const int bn = g >> 2;
    const int b  = bn >> 16;                       // NQ = 65536
    const float cy = coord[(size_t)bn * 2 + 0];
    const float cx = coord[(size_t)bn * 2 + 1];
    const float cy_ = cy + ((s & 2) ? (1.0f/64.0f) : (-1.0f/64.0f)); // vx*rx
    const float cx_ = cx + ((s & 1) ? (1.0f/64.0f) : (-1.0f/64.0f)); // vy*ry

    // hr_guide: nearest sample at unshifted coord (256x256 grid)
    const float fy = roundf(((cy + 1.0f) * 256.0f - 1.0f) * 0.5f);
    const float fx = roundf(((cx + 1.0f) * 256.0f - 1.0f) * 0.5f);
    const bool vh = (fy >= 0.f) & (fy < 256.f) & (fx >= 0.f) & (fx < 256.f);
    const int hy = min(max((int)fy, 0), 255), hx = min(max((int)fx, 0), 255);

    // feat / lr_guide / feat_coord: nearest sample at shifted coord (64x64)
    const float gy = roundf(((cy_ + 1.0f) * 64.0f - 1.0f) * 0.5f);
    const float gx = roundf(((cx_ + 1.0f) * 64.0f - 1.0f) * 0.5f);
    const bool vl = (gy >= 0.f) & (gy < 64.f) & (gx >= 0.f) & (gx < 64.f);
    const int ly = min(max((int)gy, 0), 63), lx = min(max((int)gx, 0), 63);

    unsigned short* row = bufA + r * K0PAD;
    const size_t fb = ((size_t)b * CH * 64 + ly) * 64 + lx;    // +c*4096
    const size_t hb = ((size_t)b * GD * 256 + hy) * 256 + hx;  // +c*65536
#pragma unroll 4
    for (int c = q * 16; c < q * 16 + 16; ++c) {
      const float fv = vl ? feat[fb + (size_t)c * 4096] : 0.0f;
      const float hv = vh ? hr_guide[hb + (size_t)c * 65536] : 0.0f;
      const float lv = vl ? lr_guide[fb + (size_t)c * 4096] : 0.0f;
      row[c]          = f2bf(fv);
      row[CH + c]     = f2bf(hv);
      row[2 * CH + c] = f2bf(hv - lv);
    }
    if (q == 0) {
      const float qy = vl ? (-1.0f + (2.0f * ly + 1.0f) / 64.0f) : 0.0f;
      const float qx = vl ? (-1.0f + (2.0f * lx + 1.0f) / 64.0f) : 0.0f;
      row[384] = f2bf((cy - qy) * 64.0f);
      row[385] = f2bf((cx - qx) * 64.0f);
    }
    for (int c = IN_DIM + q; c < K0PAD; c += 8) row[c] = 0;  // K padding
  }
  __syncthreads();

  // ---- fused MLP, ping-ponging LDS buffers ----
  layer_gemm(bufA, K0PAD, bufB, 1024, wpack + OFF_W0, b0, true, lane, wave);
  __syncthreads();
  layer_gemm(bufB, 1024, bufA, 512, wpack + OFF_W1, b1, true, lane, wave);
  __syncthreads();
  layer_gemm(bufA, 512, bufB, 256, wpack + OFF_W2, b2, true, lane, wave);
  __syncthreads();
  layer_gemm(bufB, 256, bufA, 128, wpack + OFF_W3, b3, true, lane, wave);
  __syncthreads();

  // ---- final 128 -> 2 layer in f32, scatter to preds[bn*8 + o*4 + s] ----
  if (t < 64) {
    const int r = t >> 1, o = t & 1;
    const unsigned short* row = bufA + r * 128;
    float acc = b4[o];
#pragma unroll 8
    for (int k = 0; k < 128; ++k) acc += bf2f(row[k]) * W4[k * 2 + o];
    const int g = blockIdx.x * TILE_M + r;
    preds[(size_t)(g >> 2) * 8 + o * 4 + (g & 3)] = acc;
  }
}

// ---------------------------------------------------------------------------
// Softmax over 4 taps (channel 1) + weighted sum (channel 0).
// ---------------------------------------------------------------------------
__global__ __launch_bounds__(256) void jiif_reduce_kernel(
    const float* __restrict__ preds, float* __restrict__ out, int total)
{
  const int i = blockIdx.x * 256 + threadIdx.x;
  if (i >= total) return;
  const float* p = preds + (size_t)i * 8;
  const float w0 = p[4], w1 = p[5], w2 = p[6], w3 = p[7];
  const float m  = fmaxf(fmaxf(w0, w1), fmaxf(w2, w3));
  const float e0 = __expf(w0 - m), e1 = __expf(w1 - m);
  const float e2 = __expf(w2 - m), e3 = __expf(w3 - m);
  const float inv = 1.0f / (e0 + e1 + e2 + e3);
  out[i] = (p[0] * e0 + p[1] * e1 + p[2] * e2 + p[3] * e3) * inv;
}

// ---------------------------------------------------------------------------
extern "C" void kernel_launch(void* const* d_in, const int* in_sizes, int n_in,
                              void* d_out, int out_size, void* d_ws, size_t ws_size,
                              hipStream_t stream)
{
  const float* feat  = (const float*)d_in[0];
  const float* coord = (const float*)d_in[1];
  const float* hr    = (const float*)d_in[2];
  const float* lr    = (const float*)d_in[3];
  const float* W0 = (const float*)d_in[4];  const float* b0 = (const float*)d_in[5];
  const float* W1 = (const float*)d_in[6];  const float* b1 = (const float*)d_in[7];
  const float* W2 = (const float*)d_in[8];  const float* b2 = (const float*)d_in[9];
  const float* W3 = (const float*)d_in[10]; const float* b3 = (const float*)d_in[11];
  const float* W4 = (const float*)d_in[12]; const float* b4 = (const float*)d_in[13];

  unsigned short* wpack = (unsigned short*)d_ws;
  float* preds = (float*)((char*)d_ws + OFF_PRED_BYTES);

  // Pack weights to fragment-major bf16 (cheap, rerun every call: deterministic).
  pack_w_kernel<<<(416 * 1024) / 256, 256, 0, stream>>>(W0, 386, 13, 1024, wpack + OFF_W0);
  pack_w_kernel<<<(1024 * 512) / 256, 256, 0, stream>>>(W1, 1024, 32, 512,  wpack + OFF_W1);
  pack_w_kernel<<<(512 * 256)  / 256, 256, 0, stream>>>(W2, 512, 16, 256,   wpack + OFF_W2);
  pack_w_kernel<<<(256 * 128)  / 256, 256, 0, stream>>>(W3, 256, 8, 128,    wpack + OFF_W3);

  (void)hipFuncSetAttribute(reinterpret_cast<const void*>(jiif_mlp_kernel),
                            hipFuncAttributeMaxDynamicSharedMemorySize, LDS_BYTES);

  // 1,048,576 rows / 32 rows-per-WG = 32768 workgroups of 8 wave32s.
  jiif_mlp_kernel<<<32768, 256, LDS_BYTES, stream>>>(
      feat, coord, hr, lr, wpack, b0, b1, b2, b3, W4, b4, preds);

  jiif_reduce_kernel<<<(4 * NQ) / 256, 256, 0, stream>>>(preds, (float*)d_out, 4 * NQ);
}
